// SelfAttention_31035433681573
// MI455X (gfx1250) — matrix operand order repaired and checked by reading, not compiled, not run on hardware
//
#include <hip/hip_runtime.h>
#include <hip/hip_bf16.h>

typedef __attribute__((ext_vector_type(16))) _Float16 v16h;
typedef __attribute__((ext_vector_type(8)))  _Float16 v8h;
typedef __attribute__((ext_vector_type(8)))  float    v8f;

#define WMMA_F16(a, b, c) \
  __builtin_amdgcn_wmma_f32_16x16x32_f16(false, (a), false, (b), (short)0, (c), false, false)

constexpr int Bn = 2;
constexpr int Ln = 2048;
constexpr int Dn = 1024;
constexpr int Hn = 16;
constexpr int DHn = 64;          // Dn / Hn
constexpr int Mrows = Bn * Ln;   // 4096
constexpr int XSTR = 36;         // f32 LDS row stride for X tiles (144 B)
constexpr int WSTR = 40;         // f16 LDS row stride for W tiles (80 B)
constexpr int KPAD = 72;         // f16 LDS row stride for K tiles (144 B)
constexpr int VSTR = 40;         // f16 LDS row stride for V^T tiles (80 B)
constexpr int PH   = 40;         // f16 row stride for P relayout buffer (80 B)

// ---------------------------------------------------------------------------
// Async global->LDS copy (16 B per lane) via CDNA5 GLOBAL_LOAD_ASYNC_TO_LDS
// inline asm (ASYNCcnt-tracked); consumers do s_wait_asynccnt 0 + barrier.
// ---------------------------------------------------------------------------
__device__ inline void async_copy16(void* lds, const void* g) {
  unsigned laddr =
      (unsigned)(unsigned long)(__attribute__((address_space(3))) void*)lds;
  asm volatile("global_load_async_to_lds_b128 %0, %1, off"
               :
               : "v"(laddr), "v"(g)
               : "memory");
}

__device__ inline void async_join() {
  asm volatile("s_wait_asynccnt 0x0" ::: "memory");
}

__device__ inline void pack8(v16h& f, int base, float4 a, float4 b) {
  f[base + 0] = (_Float16)a.x; f[base + 1] = (_Float16)a.y;
  f[base + 2] = (_Float16)a.z; f[base + 3] = (_Float16)a.w;
  f[base + 4] = (_Float16)b.x; f[base + 5] = (_Float16)b.y;
  f[base + 6] = (_Float16)b.z; f[base + 7] = (_Float16)b.w;
}

__device__ inline v16h cat8(v8h lo, v8h hi) {
  v16h r;
#pragma unroll
  for (int e = 0; e < 8; ++e) { r[e] = lo[e]; r[e + 8] = hi[e]; }
  return r;
}

// ---------------------------------------------------------------------------
// Kernel 0: one-time f32 -> f16 conversion (weights), 8 elems/thread.
// ---------------------------------------------------------------------------
__global__ __launch_bounds__(256) void to_f16(const float* __restrict__ in,
                                              _Float16* __restrict__ out, int n) {
  int i = (blockIdx.x * 256 + threadIdx.x) * 8;
  if (i >= n) return;
  const float4* p = (const float4*)(in + i);
  float4 a = p[0], b = p[1];
  v8h o;
  o[0] = (_Float16)a.x; o[1] = (_Float16)a.y; o[2] = (_Float16)a.z; o[3] = (_Float16)a.w;
  o[4] = (_Float16)b.x; o[5] = (_Float16)b.y; o[6] = (_Float16)b.z; o[7] = (_Float16)b.w;
  *(v8h*)(out + i) = o;
}

// ---------------------------------------------------------------------------
// Kernel 1: Y = scale * (X @ Wh^T), X f32, Wh f16, Y f16.
// Workgroup (8 waves) computes 128(M) x 64(N); per K-step of 32, X[128x32]
// (f32) and Wh[64x32] (f16) staged into double-buffered LDS via async copies.
// Wave w owns M-tile w, sweeps 4 N-tiles; B-frags are raw ds_load_b128 pairs.
// ---------------------------------------------------------------------------
__device__ inline void stage_x(const float* __restrict__ X, float* xs,
                               int Mbase, int kb, int tid) {
  const int srow = tid >> 3, sc = tid & 7;       // 32 rows x 8 chunks of 4 f32
#pragma unroll
  for (int i = 0; i < 4; ++i) {
    int r = srow + i * 32;
    async_copy16(&xs[r * XSTR + sc * 4],
                 X + (size_t)(Mbase + r) * Dn + kb + sc * 4);
  }
}

__device__ inline void stage_w(const _Float16* __restrict__ Wh, _Float16* ws,
                               int Nbase, int kb, int tid) {
  const int row = tid >> 2, c = tid & 3;         // 64 rows x 4 chunks of 8 f16
  async_copy16(&ws[row * WSTR + c * 8],
               Wh + (size_t)(Nbase + row) * Dn + kb + c * 8);
}

__global__ __launch_bounds__(256) void qkv_proj(const float* __restrict__ X,
                                                const _Float16* __restrict__ Wh,
                                                _Float16* __restrict__ Y,
                                                float scale) {
  __shared__ float    Xs[2][128 * XSTR];         // 36864 B
  __shared__ _Float16 Ws[2][64 * WSTR];          // 10240 B

  const int bm = blockIdx.x >> 4;                // Dn/64 = 16 N-blocks
  const int bn = blockIdx.x & 15;
  const int Mbase = bm * 128, Nbase = bn * 64;
  const int tid  = threadIdx.x;
  const int wave = tid >> 5;
  const int lane = tid & 31;
  const int m    = lane & 15;
  const int half = lane >> 4;

  v8f c[4] = {{}, {}, {}, {}};

  stage_x(X, Xs[0], Mbase, 0, tid);
  stage_w(Wh, Ws[0], Nbase, 0, tid);
  int p = 0;
  for (int kb = 0; kb < Dn; kb += 32) {
    async_join();
    __syncthreads();                             // buffer p ready for everyone
    if (kb + 32 < Dn) {                          // prefetch next into p^1
      stage_x(X, Xs[p ^ 1], Mbase, kb + 32, tid);
      stage_w(Wh, Ws[p ^ 1], Nbase, kb + 32, tid);
    }

    // A fragment: this wave's M-tile row, two 8-f32 chunks -> f16
    const float* xr = &Xs[p][(wave * 16 + m) * XSTR];
    v16h af;
    {
      const float4* p0 = (const float4*)(xr + half * 16);
      const float4* p1 = (const float4*)(xr + half * 16 + 8);
      pack8(af, 0, p0[0], p0[1]);
      pack8(af, 8, p1[0], p1[1]);
    }
#pragma unroll
    for (int nt = 0; nt < 4; ++nt) {             // B: 16 contiguous f16 of W row
      const _Float16* wr = &Ws[p][(nt * 16 + m) * WSTR + half * 16];
      v16h bf = cat8(*(const v8h*)wr, *(const v8h*)(wr + 8));
      c[nt] = WMMA_F16(af, bf, c[nt]);
    }
    p ^= 1;
  }

  const int col0  = Nbase + m;
  const int rbase = Mbase + wave * 16 + half * 8;
#pragma unroll
  for (int nt = 0; nt < 4; ++nt)
#pragma unroll
    for (int r = 0; r < 8; ++r)
      Y[(size_t)(rbase + r) * Dn + col0 + nt * 16] = (_Float16)(c[nt][r] * scale);
}

// ---------------------------------------------------------------------------
// Kernel 2: flash attention. Block = 8 waves sharing one (b,h); wave w owns a
// 16-query tile. Per 32-key chunk: K staged row-major via async copies, V
// staged TRANSPOSED (VT[d][key]) so PV B-frags are two ds_load_b128s. Online
// softmax; P relayout through per-wave f16 LDS buffer.
// ---------------------------------------------------------------------------
__device__ inline void stage_k(const _Float16* __restrict__ kh, size_t gbase,
                               _Float16* ks, int tid) {
  const int row = tid >> 3, c8 = tid & 7;        // 32 keys x 8 chunks of 8 f16
  async_copy16(&ks[row * KPAD + c8 * 8], kh + gbase + (size_t)row * Dn + c8 * 8);
}

__device__ inline void stage_v_t(const _Float16* __restrict__ vh, size_t gbase,
                                 _Float16* vt, int tid) {
  const int key = tid >> 3, c8 = tid & 7;        // coalesced read, scattered store
  v8h val = *(const v8h*)(vh + gbase + (size_t)key * Dn + c8 * 8);
#pragma unroll
  for (int j = 0; j < 8; ++j)
    vt[(c8 * 8 + j) * VSTR + key] = val[j];
}

__global__ __launch_bounds__(256) void attn(const _Float16* __restrict__ qh,
                                            const _Float16* __restrict__ kh,
                                            const _Float16* __restrict__ vh,
                                            const int* __restrict__ qmask,
                                            float* __restrict__ out) {
  __shared__ _Float16 Ks[2][32 * KPAD];          // 9216 B
  __shared__ _Float16 VT[2][64 * VSTR];          // 10240 B
  __shared__ _Float16 Pbuf[8][16 * PH];          // 10240 B

  const int tid  = threadIdx.x;
  const int wave = tid >> 5;
  const int lane = tid & 31;
  const int gt   = blockIdx.x * 8 + wave;        // q-tile id; 8 per block share (b,h)
  const int b  = gt >> 11;
  const int rm = gt & 2047;
  const int h  = rm >> 7;
  const int qt = rm & 127;
  const int m    = lane & 15;
  const int half = lane >> 4;
  _Float16* P = &Pbuf[wave][0];
  const size_t bL = (size_t)b * Ln;
  const size_t kvbase = bL * Dn + (size_t)h * DHn;

  // ---- Q as two A-fragments (K = DH = 64) ----
  const _Float16* __restrict__ qp =
      qh + (size_t)(bL + qt * 16 + m) * Dn + h * DHn;
  v16h aq[2];
#pragma unroll
  for (int f = 0; f < 2; ++f)
    aq[f] = cat8(*(const v8h*)(qp + f * 32 + half * 16),
                 *(const v8h*)(qp + f * 32 + half * 16 + 8));

  float mrun[8], lrun[8];
  v8f o[4] = {{}, {}, {}, {}};
#pragma unroll
  for (int r = 0; r < 8; ++r) { mrun[r] = -1e30f; lrun[r] = 0.0f; }

  stage_k(kh, kvbase, Ks[0], tid);
  stage_v_t(vh, kvbase, VT[0], tid);
  int p = 0;
  for (int kc = 0; kc < Ln; kc += 32) {
    async_join();
    __syncthreads();                             // buffer p holds keys kc..kc+31
    if (kc + 32 < Ln) {
      size_t gb = kvbase + (size_t)(kc + 32) * Dn;
      stage_k(kh, gb, Ks[p ^ 1], tid);
      stage_v_t(vh, gb, VT[p ^ 1], tid);
    }

    // ---- S = Q @ K^T : two 16-key column tiles, K-dim 64 ----
    const _Float16* kp0 = &Ks[p][m * KPAD];          // key kc + m
    const _Float16* kp1 = &Ks[p][(16 + m) * KPAD];   // key kc + 16 + m
    v8f s0 = {}, s1 = {};
#pragma unroll
    for (int f = 0; f < 2; ++f) {
      const int d0 = f * 32 + half * 16;
      v16h bk0 = cat8(*(const v8h*)(kp0 + d0), *(const v8h*)(kp0 + d0 + 8));
      v16h bk1 = cat8(*(const v8h*)(kp1 + d0), *(const v8h*)(kp1 + d0 + 8));
      s0 = WMMA_F16(aq[f], bk0, s0);
      s1 = WMMA_F16(aq[f], bk1, s1);
    }

    const bool v0 = qmask[bL + kc + m] != 0;
    const bool v1 = qmask[bL + kc + 16 + m] != 0;

    // ---- online softmax ----
    float cmax[8];
#pragma unroll
    for (int r = 0; r < 8; ++r)
      cmax[r] = fmaxf(v0 ? s0[r] : -1e30f, v1 ? s1[r] : -1e30f);
#pragma unroll
    for (int xm = 8; xm >= 1; xm >>= 1)
#pragma unroll
      for (int r = 0; r < 8; ++r)
        cmax[r] = fmaxf(cmax[r], __shfl_xor(cmax[r], xm, 32));

    float p0[8], p1[8], csum[8], scl[8];
#pragma unroll
    for (int r = 0; r < 8; ++r) {
      float mn = fmaxf(mrun[r], cmax[r]);
      scl[r]  = __expf(mrun[r] - mn);
      mrun[r] = mn;
      p0[r] = v0 ? __expf(s0[r] - mn) : 0.0f;
      p1[r] = v1 ? __expf(s1[r] - mn) : 0.0f;
      csum[r] = p0[r] + p1[r];
    }
#pragma unroll
    for (int xm = 8; xm >= 1; xm >>= 1)
#pragma unroll
      for (int r = 0; r < 8; ++r)
        csum[r] += __shfl_xor(csum[r], xm, 32);
#pragma unroll
    for (int r = 0; r < 8; ++r) {
      lrun[r] = lrun[r] * scl[r] + csum[r];
      o[0][r] *= scl[r]; o[1][r] *= scl[r];
      o[2][r] *= scl[r]; o[3][r] *= scl[r];
    }

    // ---- P: C-layout -> per-wave f16 LDS -> A-layout ----
#pragma unroll
    for (int r = 0; r < 8; ++r) {
      int mm = r + 8 * half;
      P[mm * PH + m]      = (_Float16)p0[r];
      P[mm * PH + 16 + m] = (_Float16)p1[r];
    }
    asm volatile("" ::: "memory");               // same-wave DS ops are in-order
    v16h ap = cat8(*(const v8h*)(P + m * PH + half * 16),
                   *(const v8h*)(P + m * PH + half * 16 + 8));

    // ---- O += P @ V : 4 d-tiles, K = 32 keys (from transposed V) ----
#pragma unroll
    for (int t = 0; t < 4; ++t) {
      const _Float16* vr = &VT[p][(t * 16 + m) * VSTR + half * 16];
      v16h bv = cat8(*(const v8h*)vr, *(const v8h*)(vr + 8));
      o[t] = WMMA_F16(ap, bv, o[t]);
    }
    p ^= 1;
  }

  // ---- normalize + store f32 ----
#pragma unroll
  for (int r = 0; r < 8; ++r) {
    float inv = 1.0f / (lrun[r] + 1e-13f);
    size_t row = bL + qt * 16 + r + 8 * half;
#pragma unroll
    for (int t = 0; t < 4; ++t)
      out[row * Dn + h * DHn + t * 16 + m] = o[t][r] * inv;
  }
}

extern "C" void kernel_launch(void* const* d_in, const int* in_sizes, int n_in,
                              void* d_out, int out_size, void* d_ws, size_t ws_size,
                              hipStream_t stream) {
  const float* queries = (const float*)d_in[0];
  const int*   qmask   = (const int*)d_in[1];
  const float* keys    = (const float*)d_in[2];
  const float* values  = (const float*)d_in[3];
  const float* Wq      = (const float*)d_in[4];
  const float* Wk      = (const float*)d_in[5];
  const float* Wv      = (const float*)d_in[6];
  float* out = (float*)d_out;

  // workspace: qh/kh/vh f16 [B*L, D] + f16 weights [D, D] x3  (30 MB)
  _Float16* qh  = (_Float16*)d_ws;
  _Float16* kh  = qh + (size_t)Mrows * Dn;
  _Float16* vh  = kh + (size_t)Mrows * Dn;
  _Float16* wqh = vh + (size_t)Mrows * Dn;
  _Float16* wkh = wqh + (size_t)Dn * Dn;
  _Float16* wvh = wkh + (size_t)Dn * Dn;

  dim3 blk(256);
  const int nw = Dn * Dn;                        // 1M elems per weight
  dim3 grdC(nw / (256 * 8));                     // 512 blocks
  to_f16<<<grdC, blk, 0, stream>>>(Wq, wqh, nw);
  to_f16<<<grdC, blk, 0, stream>>>(Wk, wkh, nw);
  to_f16<<<grdC, blk, 0, stream>>>(Wv, wvh, nw);

  dim3 grdP((Mrows / 128) * (Dn / 64));          // 512 blocks
  qkv_proj<<<grdP, blk, 0, stream>>>(queries, wqh, qh, 0.125f);  // DH^-0.5
  qkv_proj<<<grdP, blk, 0, stream>>>(keys,    wkh, kh, 1.0f);
  qkv_proj<<<grdP, blk, 0, stream>>>(values,  wvh, vh, 1.0f);

  dim3 grdA((Bn * Hn * (Ln / 16)) / 8);          // 512 blocks
  attn<<<grdA, blk, 0, stream>>>(qh, kh, vh, qmask, out);
}
